// MultiHeadAttention_59047210386004
// MI455X (gfx1250) — compile-verified, matrix operand order checked by reading, not run on hardware
//
#include <hip/hip_runtime.h>
#include <hip/hip_bf16.h>
#include <stdint.h>

// ---------------------------------------------------------------------------
// MHA forward for MI455X (gfx1250): all GEMMs via v_wmma_f32_16x16x32_f16.
// Layout convention for every GEMM:  C[M,N] = A[M,K] (row) x Bt[N,K] (row),
// so A and B fragments use the identical loader that matches the
// CDNA5 16-bit A-matrix VGPR layout (ISA 7.12.2).
// mha_proj additionally stages its block-shared A tile through LDS with
// global_load_async_to_lds_b64 (ASYNCcnt path), double-buffered.
// ---------------------------------------------------------------------------

typedef _Float16 v16h __attribute__((ext_vector_type(16)));
typedef _Float16 v8h  __attribute__((ext_vector_type(8)));
typedef float    v8f  __attribute__((ext_vector_type(8)));

#define B_   2
#define S_   2048
#define D_   1024
#define H_   16
#define DH_  64
#define BS_  4096

#define WMMA16(a, b, c) \
  __builtin_amdgcn_wmma_f32_16x16x32_f16(false, (a), false, (b), (short)0, (c), false, false)

// A/B fragment loader, f16 source, row-major [rows, ld], 16x32 tile at (row0,k0).
// lane 0-15: rows row0..row0+15, K-half 0 ; lane 16-31: rows, K-half 1.
// halves 0..7  <- K = kh*8 + 0..7 ; halves 8..15 <- K = 16 + kh*8 + 0..7
__device__ __forceinline__ v16h ldfrag_h(const _Float16* __restrict__ base,
                                         int row0, int k0, int ld, int lane) {
  const int r  = row0 + (lane & 15);
  const int kh = (lane >> 4) << 3;
  const _Float16* p = base + (size_t)r * ld + k0 + kh;
  union { v16h v; v8h h[2]; } f;
  f.h[0] = *(const v8h*)(p);
  f.h[1] = *(const v8h*)(p + 16);
  return f.v;
}

// Same fragment from LDS (generic pointer into __shared__) -> ds_load_b128 x2.
__device__ __forceinline__ v16h ldfrag_lds(const _Float16* base, int row0,
                                           int ld, int lane) {
  const int r  = row0 + (lane & 15);
  const int kh = (lane >> 4) << 3;
  const _Float16* p = base + r * ld + kh;
  union { v16h v; v8h h[2]; } f;
  f.h[0] = *(const v8h*)(p);
  f.h[1] = *(const v8h*)(p + 16);
  return f.v;
}

// Fragment converting from f32 source on the fly (used for attn rows).
__device__ __forceinline__ v16h ldfrag_f(const float* __restrict__ base,
                                         int row0, int k0, int ld, int lane) {
  const int r  = row0 + (lane & 15);
  const int kh = (lane >> 4) << 3;
  const float* p = base + (size_t)r * ld + k0 + kh;
  v16h v;
#pragma unroll
  for (int i = 0; i < 8; ++i) {
    v[i]     = (_Float16)p[i];
    v[i + 8] = (_Float16)p[i + 16];
  }
  return v;
}

// Issue all 8 WMMAs of a 32x64 wave tile back-to-back.
__device__ __forceinline__ void mma_8(v8f c[2][4], v16h a0, v16h a1,
                                      v16h b0, v16h b1, v16h b2, v16h b3) {
  c[0][0] = WMMA16(a0, b0, c[0][0]);
  c[1][0] = WMMA16(a1, b0, c[1][0]);
  c[0][1] = WMMA16(a0, b1, c[0][1]);
  c[1][1] = WMMA16(a1, b1, c[1][1]);
  c[0][2] = WMMA16(a0, b2, c[0][2]);
  c[1][2] = WMMA16(a1, b2, c[1][2]);
  c[0][3] = WMMA16(a0, b3, c[0][3]);
  c[1][3] = WMMA16(a1, b3, c[1][3]);
}

// ---------------------------------------------------------------------------
// Elementwise f32 -> f16 convert
// ---------------------------------------------------------------------------
__global__ void mha_cvt16(const float* __restrict__ x, _Float16* __restrict__ y, int n) {
  int i = blockIdx.x * blockDim.x + threadIdx.x;
  if (i < n) y[i] = (_Float16)x[i];
}

// Weight transpose + convert: Wt[n][k] = (f16) W[k][n]   (1024 x 1024)
__global__ void mha_cvtWt(const float* __restrict__ W, _Float16* __restrict__ Wt) {
  int kk = blockIdx.x * 16 + threadIdx.x;
  int n  = blockIdx.y * 16 + threadIdx.y;
  Wt[(size_t)n * D_ + kk] = (_Float16)W[(size_t)kk * D_ + n];
}

// ---------------------------------------------------------------------------
// Projection GEMM: X[4096,1024]f16 x Wt[1024,1024]f16 + bias -> f16 heads.
// All 8 waves of a block share one 32-row A tile -> staged via LDS using
// async global->LDS copies (double buffered). B tiles direct from L2.
// store_vt == 0 : out is [B,H,S,DH] (q,k) ; 1 : out is [B,H,DH,S] (v^T)
// ---------------------------------------------------------------------------
__global__ void mha_proj(const _Float16* __restrict__ X,
                         const _Float16* __restrict__ Wt,
                         const float* __restrict__ bias,
                         _Float16* __restrict__ out, int store_vt) {
  __shared__ _Float16 As[2][32][32];            // 2 x 2KB ping-pong
  const int t    = threadIdx.x;
  const int lane = t & 31;
  const int widx = t >> 5;
  const int m0 = (blockIdx.x >> 1) * 32;                 // uniform per block
  const int n0 = (((blockIdx.x & 1) << 3) + widx) * 64;  // per-wave N tile

  const int srow = t >> 3;            // 0..31 : row staged by this thread
  const int scol = (t & 7) * 4;       // 4 halves = 8 bytes per thread

  // prologue: stage k0 = 0 into buffer 0
  {
    const _Float16* gp = X + (size_t)(m0 + srow) * D_ + scol;
    uint32_t laddr = (uint32_t)(uintptr_t)&As[0][srow][scol];
    asm volatile("global_load_async_to_lds_b64 %0, %1, off"
                 :: "v"(laddr), "v"(gp) : "memory");
  }

  v8f c[2][4] = {};
  int buf = 0;
  for (int k0 = 0; k0 < D_; k0 += 32) {
    asm volatile("s_wait_asynccnt 0x0" ::: "memory");
    __syncthreads();                                   // tile[buf] ready; tile[buf^1] free
    if (k0 + 32 < D_) {                                // stage next tile (uniform branch)
      const _Float16* gp = X + (size_t)(m0 + srow) * D_ + (k0 + 32) + scol;
      uint32_t laddr = (uint32_t)(uintptr_t)&As[buf ^ 1][srow][scol];
      asm volatile("global_load_async_to_lds_b64 %0, %1, off"
                   :: "v"(laddr), "v"(gp) : "memory");
    }
    v16h a0 = ldfrag_lds(&As[buf][0][0],  0, 32, lane);
    v16h a1 = ldfrag_lds(&As[buf][0][0], 16, 32, lane);
    v16h b0 = ldfrag_h(Wt, n0 +  0, k0, D_, lane);
    v16h b1 = ldfrag_h(Wt, n0 + 16, k0, D_, lane);
    v16h b2 = ldfrag_h(Wt, n0 + 32, k0, D_, lane);
    v16h b3 = ldfrag_h(Wt, n0 + 48, k0, D_, lane);
    mma_8(c, a0, a1, b0, b1, b2, b3);
    buf ^= 1;
  }

  // D layout: VGPR r -> row r + 8*(lane>=16), col = lane&15
  const int colb = lane & 15;
  const int rofs = (lane >> 4) << 3;
#pragma unroll
  for (int mi = 0; mi < 2; ++mi) {
#pragma unroll
    for (int j = 0; j < 4; ++j) {
      const int col = n0 + j * 16 + colb;
      const float bv = bias[col];
      const int h = col >> 6, d = col & 63;
#pragma unroll
      for (int r = 0; r < 8; ++r) {
        const int row = m0 + mi * 16 + rofs + r;       // row = b*S + s
        const int bb = row >> 11, s = row & (S_ - 1);
        const float val = c[mi][j][r] + bv;
        size_t idx = store_vt
          ? ((((size_t)bb * H_ + h) * DH_ + d) * S_ + s)
          : ((((size_t)bb * H_ + h) * S_ + s) * DH_ + d);
        out[idx] = (_Float16)val;
      }
    }
  }
}

// ---------------------------------------------------------------------------
// Scores: per (b,h)  q[S,DH] x k[S,DH]^T * 1/sqrt(DH)  -> attn f32 [S,S]
// ---------------------------------------------------------------------------
__global__ void mha_scores(const _Float16* __restrict__ qh,
                           const _Float16* __restrict__ kh,
                           float* __restrict__ attn) {
  const int bh = blockIdx.y;
  const _Float16* q = qh + (size_t)bh * S_ * DH_;
  const _Float16* k = kh + (size_t)bh * S_ * DH_;
  float* out = attn + (size_t)bh * S_ * S_;

  const int lane = threadIdx.x & 31;
  const int wave = blockIdx.x * (blockDim.x >> 5) + (threadIdx.x >> 5);
  const int tn = wave & 31;        // 2048/64 = 32 N tiles
  const int tm = wave >> 5;        // 2048/32 = 64 M tiles
  const int m0 = tm * 32, n0 = tn * 64;

  v8f c[2][4] = {};
#pragma unroll
  for (int k0 = 0; k0 < DH_; k0 += 32) {
    v16h a0 = ldfrag_h(q, m0,      k0, DH_, lane);
    v16h a1 = ldfrag_h(q, m0 + 16, k0, DH_, lane);
    v16h b0 = ldfrag_h(k, n0 +  0, k0, DH_, lane);
    v16h b1 = ldfrag_h(k, n0 + 16, k0, DH_, lane);
    v16h b2 = ldfrag_h(k, n0 + 32, k0, DH_, lane);
    v16h b3 = ldfrag_h(k, n0 + 48, k0, DH_, lane);
    mma_8(c, a0, a1, b0, b1, b2, b3);
  }
  const float scale = 0.125f;      // 1/sqrt(64)
  const int colb = lane & 15;
  const int rofs = (lane >> 4) << 3;
#pragma unroll
  for (int mi = 0; mi < 2; ++mi)
#pragma unroll
    for (int j = 0; j < 4; ++j) {
      const int col = n0 + j * 16 + colb;
#pragma unroll
      for (int r = 0; r < 8; ++r) {
        const int row = m0 + mi * 16 + rofs + r;
        out[(size_t)row * S_ + col] = c[mi][j][r] * scale;
      }
    }
}

// ---------------------------------------------------------------------------
// Row softmax over 2048 cols, in place. One 256-thread block per row.
// ---------------------------------------------------------------------------
__global__ void mha_softmax(float* __restrict__ attn) {
  float* p = attn + (size_t)blockIdx.x * S_;
  const int t = threadIdx.x;
  const int lane = t & 31, wid = t >> 5;
  __shared__ float redm[8];
  __shared__ float reds[8];

  float x[8];
  float m = -3.402823466e38f;
#pragma unroll
  for (int i = 0; i < 8; ++i) { x[i] = p[t + i * 256]; m = fmaxf(m, x[i]); }
#pragma unroll
  for (int o = 16; o > 0; o >>= 1) m = fmaxf(m, __shfl_xor(m, o, 32));
  if (lane == 0) redm[wid] = m;
  __syncthreads();
  float mt = redm[0];
#pragma unroll
  for (int w = 1; w < 8; ++w) mt = fmaxf(mt, redm[w]);

  float s = 0.f;
#pragma unroll
  for (int i = 0; i < 8; ++i) { x[i] = __expf(x[i] - mt); s += x[i]; }
#pragma unroll
  for (int o = 16; o > 0; o >>= 1) s += __shfl_xor(s, o, 32);
  if (lane == 0) reds[wid] = s;
  __syncthreads();
  float st = 0.f;
#pragma unroll
  for (int w = 0; w < 8; ++w) st += reds[w];
  const float inv = 1.0f / st;
#pragma unroll
  for (int i = 0; i < 8; ++i) p[t + i * 256] = x[i] * inv;
}

// ---------------------------------------------------------------------------
// attn @ V : per (b,h)  attn[S,S]f32 x Vt[DH,S]f16 -> ctx f16 [B,S,H*DH]
// A fragments converted f32->f16 on the fly. K = 2048.
// ---------------------------------------------------------------------------
__global__ void mha_attnv(const float* __restrict__ attn,
                          const _Float16* __restrict__ vt,
                          _Float16* __restrict__ ctx) {
  const int bh = blockIdx.y;
  const int bb = bh >> 4, h = bh & 15;
  const float* a = attn + (size_t)bh * S_ * S_;
  const _Float16* v = vt + (size_t)bh * DH_ * S_;

  const int lane = threadIdx.x & 31;
  const int wave = blockIdx.x * (blockDim.x >> 5) + (threadIdx.x >> 5);
  const int m0 = wave * 32;        // 64 waves cover M=2048; N tile = full 64

  v8f c[2][4] = {};
  for (int k0 = 0; k0 < S_; k0 += 32) {
    if (k0 + 32 < S_) {            // prefetch next A lines (global_prefetch_b8)
      __builtin_prefetch(a + (size_t)(m0 + (lane & 15)) * S_ + k0 + 32, 0, 1);
      __builtin_prefetch(a + (size_t)(m0 + 16 + (lane & 15)) * S_ + k0 + 32, 0, 1);
    }
    v16h a0 = ldfrag_f(a, m0,      k0, S_, lane);
    v16h a1 = ldfrag_f(a, m0 + 16, k0, S_, lane);
    v16h b0 = ldfrag_h(v,  0, k0, S_, lane);
    v16h b1 = ldfrag_h(v, 16, k0, S_, lane);
    v16h b2 = ldfrag_h(v, 32, k0, S_, lane);
    v16h b3 = ldfrag_h(v, 48, k0, S_, lane);
    mma_8(c, a0, a1, b0, b1, b2, b3);
  }
  const int colb = lane & 15;
  const int rofs = (lane >> 4) << 3;
#pragma unroll
  for (int mi = 0; mi < 2; ++mi)
#pragma unroll
    for (int j = 0; j < 4; ++j) {
      const int d = j * 16 + colb;
#pragma unroll
      for (int r = 0; r < 8; ++r) {
        const int s = m0 + mi * 16 + rofs + r;
        ctx[(((size_t)bb * S_ + s) * D_) + h * DH_ + d] = (_Float16)c[mi][j][r];
      }
    }
}

// ---------------------------------------------------------------------------
// Output projection: ctx[4096,1024]f16 x Wot[1024,1024]f16 + bo -> f32 d_out
// ---------------------------------------------------------------------------
__global__ void mha_oproj(const _Float16* __restrict__ ctx,
                          const _Float16* __restrict__ Wt,
                          const float* __restrict__ bias,
                          float* __restrict__ out) {
  const int lane = threadIdx.x & 31;
  const int wave = blockIdx.x * (blockDim.x >> 5) + (threadIdx.x >> 5);
  const int tn = wave & 15;
  const int tm = wave >> 4;
  const int m0 = tm * 32, n0 = tn * 64;

  v8f c[2][4] = {};
  for (int k0 = 0; k0 < D_; k0 += 32) {
    if (k0 + 32 < D_) {
      __builtin_prefetch(ctx + (size_t)(m0 + (lane & 15)) * D_ + k0 + 32, 0, 1);
    }
    v16h a0 = ldfrag_h(ctx, m0,      k0, D_, lane);
    v16h a1 = ldfrag_h(ctx, m0 + 16, k0, D_, lane);
    v16h b0 = ldfrag_h(Wt, n0 +  0, k0, D_, lane);
    v16h b1 = ldfrag_h(Wt, n0 + 16, k0, D_, lane);
    v16h b2 = ldfrag_h(Wt, n0 + 32, k0, D_, lane);
    v16h b3 = ldfrag_h(Wt, n0 + 48, k0, D_, lane);
    mma_8(c, a0, a1, b0, b1, b2, b3);
  }
  const int colb = lane & 15;
  const int rofs = (lane >> 4) << 3;
#pragma unroll
  for (int mi = 0; mi < 2; ++mi)
#pragma unroll
    for (int j = 0; j < 4; ++j) {
      const int col = n0 + j * 16 + colb;
      const float bv = bias[col];
#pragma unroll
      for (int r = 0; r < 8; ++r) {
        const int row = m0 + mi * 16 + rofs + r;
        out[(size_t)row * D_ + col] = c[mi][j][r] + bv;
      }
    }
}

// ---------------------------------------------------------------------------
extern "C" void kernel_launch(void* const* d_in, const int* in_sizes, int n_in,
                              void* d_out, int out_size, void* d_ws, size_t ws_size,
                              hipStream_t stream) {
  const float* Q  = (const float*)d_in[0];
  const float* K  = (const float*)d_in[1];
  const float* V  = (const float*)d_in[2];
  const float* Wq = (const float*)d_in[3];
  const float* bq = (const float*)d_in[4];
  const float* Wk = (const float*)d_in[5];
  const float* bk = (const float*)d_in[6];
  const float* Wv = (const float*)d_in[7];
  const float* bv = (const float*)d_in[8];
  const float* Wo = (const float*)d_in[9];
  const float* bo = (const float*)d_in[10];

  float* out  = (float*)d_out;                       // [B,S,D]  = 4,194,304
  float* attn = (float*)d_out + (size_t)BS_ * D_;    // [B,H,S,S] f32

  // workspace layout (halves)
  _Float16* w = (_Float16*)d_ws;
  const size_t NX = (size_t)BS_ * D_;   // 4,194,304
  const size_t NW = (size_t)D_ * D_;    // 1,048,576
  _Float16* Xq  = w;              // converted inputs
  _Float16* Xk  = Xq + NX;
  _Float16* Xv  = Xk + NX;
  _Float16* Wqt = Xv + NX;        // transposed f16 weights
  _Float16* Wkt = Wqt + NW;
  _Float16* Wvt = Wkt + NW;
  _Float16* Wot = Wvt + NW;
  _Float16* qh  = Wot + NW;       // [B,H,S,DH]
  _Float16* kh  = qh + NX;        // [B,H,S,DH]
  _Float16* vt  = kh + NX;        // [B,H,DH,S]
  _Float16* ctx = vt + NX;        // [B,S,D]

  // 1) convert activations and weights to f16 (weights transposed)
  {
    int n = (int)NX;
    int blocks = (n + 255) / 256;
    mha_cvt16<<<blocks, 256, 0, stream>>>(Q, Xq, n);
    mha_cvt16<<<blocks, 256, 0, stream>>>(K, Xk, n);
    mha_cvt16<<<blocks, 256, 0, stream>>>(V, Xv, n);
    dim3 g(64, 64), b(16, 16);
    mha_cvtWt<<<g, b, 0, stream>>>(Wq, Wqt);
    mha_cvtWt<<<g, b, 0, stream>>>(Wk, Wkt);
    mha_cvtWt<<<g, b, 0, stream>>>(Wv, Wvt);
    mha_cvtWt<<<g, b, 0, stream>>>(Wo, Wot);
  }

  // 2) projections: 2048 waves = 256 blocks x 8 waves
  mha_proj<<<256, 256, 0, stream>>>(Xq, Wqt, bq, qh, 0);
  mha_proj<<<256, 256, 0, stream>>>(Xk, Wkt, bk, kh, 0);
  mha_proj<<<256, 256, 0, stream>>>(Xv, Wvt, bv, vt, 1);

  // 3) scores: per head 2048 waves -> grid (256, 32)
  mha_scores<<<dim3(256, B_ * H_), 256, 0, stream>>>(qh, kh, attn);

  // 4) softmax over 65536 rows
  mha_softmax<<<B_ * H_ * S_, 256, 0, stream>>>(attn);

  // 5) attn @ V : per head 64 waves -> grid (8, 32)
  mha_attnv<<<dim3(8, B_ * H_), 256, 0, stream>>>(attn, vt, ctx);

  // 6) output projection
  mha_oproj<<<256, 256, 0, stream>>>(ctx, Wot, bo, out);
}
